// GraphCast_77532749627490
// MI455X (gfx1250) — matrix operand order repaired and checked by reading, not compiled, not run on hardware
//
#include <hip/hip_runtime.h>
#include <hip/hip_bf16.h>
#include <math.h>

#define B_  4
#define NG_ 40962
#define NM_ 10242
#define E_  65536
#define LD_ 256
#define NL_ 6

typedef __attribute__((ext_vector_type(16))) __bf16 v16bf;
typedef __attribute__((ext_vector_type(8)))  float  v8f;

enum { EPI_BF16 = 0, EPI_LNSILU = 1, EPI_X = 2 };

__device__ __forceinline__ unsigned short f2bf(float f) {
  unsigned int x = __float_as_uint(f);
  unsigned int r = x + 0x7FFFu + ((x >> 16) & 1u);   // round-to-nearest-even
  return (unsigned short)(r >> 16);
}
__device__ __forceinline__ unsigned int f2bf2(float lo, float hi) {
  return (unsigned int)f2bf(lo) | ((unsigned int)f2bf(hi) << 16);
}
__device__ __forceinline__ float bf2f(unsigned short u) {
  return __uint_as_float(((unsigned int)u) << 16);
}
__device__ __forceinline__ float waveSum(float v) {
#pragma unroll
  for (int m = 1; m < 32; m <<= 1) v += __shfl_xor(v, m, 32);
  return v;
}
__device__ __forceinline__ float siluf(float y) { return y / (1.0f + expf(-y)); }

__device__ __forceinline__ v8f wmma_bf16(const v16bf& a, const v16bf& b, v8f c) {
  return __builtin_amdgcn_wmma_f32_16x16x32_bf16(false, a, false, b,
                                                 (short)0, c, false, false);
}
__device__ __forceinline__ void ld_frag_a(v16bf* f, const unsigned short* p) {
  ((uint4*)f)[0] = *(const uint4*)(p);
  ((uint4*)f)[1] = *(const uint4*)(p + 16);
}
__device__ __forceinline__ void ld_frag_b(v16bf* f, const unsigned short* p) {
  ((uint4*)f)[0] = *(const uint4*)(p);
  ((uint4*)f)[1] = *(const uint4*)(p + 16);
}

// ---------------------------------------------------------------------------
// One-time weight convert+transpose: W_f32[K x 256] -> Wt_bf16[256 x KP],
// zero-padded so GEMM loops need no bounds checks.
// ---------------------------------------------------------------------------
__global__ __launch_bounds__(256) void wt_convert_kernel(
    const float* __restrict__ W, unsigned short* __restrict__ Wt,
    int K, int KP, long total /* = 256*KP/2 */)
{
  long i = (long)blockIdx.x * 256 + threadIdx.x;
  if (i >= total) return;
  int c  = (int)(i & 255);
  int kk = (int)(i >> 8) * 2;
  float w0 = (kk     < K) ? W[(size_t)kk * 256 + c]       : 0.f;
  float w1 = (kk + 1 < K) ? W[(size_t)(kk + 1) * 256 + c] : 0.f;
  *(unsigned int*)(Wt + (size_t)c * KP + kk) = f2bf2(w0, w1);
}

// ---------------------------------------------------------------------------
// Encoder stage 1: h = silu(ln(grid_input @ w1 + b1))  (K=2, VALU)
// ---------------------------------------------------------------------------
__global__ __launch_bounds__(256) void encode_kernel(
    const float* __restrict__ gin, const float* __restrict__ w1,
    const float* __restrict__ b1, const float* __restrict__ g,
    const float* __restrict__ be, unsigned short* __restrict__ out, int R)
{
  const int wave = threadIdx.x >> 5, lane = threadIdx.x & 31;
  long r = (long)blockIdx.x * 8 + wave;
  if (r >= R) return;
  float i0 = gin[r * 2], i1 = gin[r * 2 + 1];
  float v[8], s = 0.f, s2 = 0.f;
#pragma unroll
  for (int i = 0; i < 8; ++i) {
    int c = lane * 8 + i;
    float t = i0 * w1[c] + i1 * w1[256 + c] + b1[c];
    v[i] = t; s += t; s2 += t * t;
  }
  s = waveSum(s); s2 = waveSum(s2);
  float mu = s * (1.f / 256.f);
  float var = s2 * (1.f / 256.f) - mu * mu;
  float rs = rsqrtf(var + 1e-5f);
  uint4 o; unsigned short* ou = (unsigned short*)&o;
#pragma unroll
  for (int i = 0; i < 8; ++i) {
    int c = lane * 8 + i;
    float y = (v[i] - mu) * rs * g[c] + be[c];
    ou[i] = f2bf(siluf(y));
  }
  ((uint4*)(out + (size_t)r * 256))[lane] = o;
}

// ---------------------------------------------------------------------------
// Generic WMMA GEMM: out[R x 256] = A_bf16[R x KP] @ Wt_bf16^T + bias.
// 256 threads = 8 waves; each wave computes a 2x2 grid of 16x16 tiles
// (rows {row0, row0+16}, cols {w*16, w*16+128}): 8 global b128 loads feed
// 4 v_wmma per 32-K chunk (A and B fragments each reused twice).
// Barrier-free main loop; LDS + 1 barrier only in the epilogue.
// ---------------------------------------------------------------------------
template <int KP, int MODE>
__global__ __launch_bounds__(256) void gemm256_kernel(
    const unsigned short* __restrict__ A,    // [R x KP] bf16 (zero-padded)
    const unsigned short* __restrict__ Wt,   // [256 x KP] bf16 (zero-padded)
    const float* __restrict__ bias, const float* __restrict__ gamma,
    const float* __restrict__ beta, const float* __restrict__ resid,
    float* __restrict__ outF, unsigned short* __restrict__ outH, int R)
{
  __shared__ float sOut[32 * 256];                     // 32 KB
  const int tid  = threadIdx.x;
  const int wave = tid >> 5, lane = tid & 31;
  const int half = lane >> 4, l16 = lane & 15;
  const int n0 = wave * 16 + l16;
  const int n1 = n0 + 128;
  const long row0 = (long)blockIdx.x * 32;

  long r0 = row0 + l16;       if (r0 >= R) r0 = R - 1;  // clamp (guarded stores)
  long r1 = row0 + 16 + l16;  if (r1 >= R) r1 = R - 1;
  const unsigned short* aRow0 = A  + (size_t)r0 * KP + half * 8;
  const unsigned short* aRow1 = A  + (size_t)r1 * KP + half * 8;
  const unsigned short* wRow0 = Wt + (size_t)n0 * KP + half * 8;
  const unsigned short* wRow1 = Wt + (size_t)n1 * KP + half * 8;

  v8f a00 = {0.f,0.f,0.f,0.f,0.f,0.f,0.f,0.f};
  v8f a01 = a00, a10 = a00, a11 = a00;
#pragma unroll
  for (int kc = 0; kc < KP; kc += 32) {
    v16bf av0, av1, bv0, bv1;
    ld_frag_a(&av0, aRow0 + kc);
    ld_frag_a(&av1, aRow1 + kc);
    ld_frag_b(&bv0, wRow0 + kc);
    ld_frag_b(&bv1, wRow1 + kc);
    a00 = wmma_bf16(av0, bv0, a00);
    a01 = wmma_bf16(av0, bv1, a01);
    a10 = wmma_bf16(av1, bv0, a10);
    a11 = wmma_bf16(av1, bv1, a11);
  }
  const float b0 = bias[n0], b1v = bias[n1];
#pragma unroll
  for (int r = 0; r < 8; ++r) {
    int rr0 = (r + half * 8) * 256;
    int rr1 = (16 + r + half * 8) * 256;
    sOut[rr0 + n0] = a00[r] + b0;
    sOut[rr0 + n1] = a01[r] + b1v;
    sOut[rr1 + n0] = a10[r] + b0;
    sOut[rr1 + n1] = a11[r] + b1v;
  }
  __syncthreads();

#pragma unroll
  for (int j = 0; j < 4; ++j) {
    int row = wave + 8 * j;
    long gr = row0 + row;
    if (gr >= R) continue;
    float v[8];
#pragma unroll
    for (int i = 0; i < 8; ++i) v[i] = sOut[row * 256 + lane * 8 + i];

    if (MODE == EPI_LNSILU) {
      float s = 0.f, s2 = 0.f;
#pragma unroll
      for (int i = 0; i < 8; ++i) { s += v[i]; s2 += v[i] * v[i]; }
      s = waveSum(s); s2 = waveSum(s2);
      float mu = s * (1.f / 256.f);
      float var = s2 * (1.f / 256.f) - mu * mu;
      float rs = rsqrtf(var + 1e-5f);
      uint4 o; unsigned short* ou = (unsigned short*)&o;
#pragma unroll
      for (int i = 0; i < 8; ++i) {
        int c = lane * 8 + i;
        float y = (v[i] - mu) * rs * gamma[c] + beta[c];
        ou[i] = f2bf(siluf(y));
      }
      ((uint4*)(outH + (size_t)gr * 256))[lane] = o;
    } else if (MODE == EPI_X) {
      float rv[8] = {0.f,0.f,0.f,0.f,0.f,0.f,0.f,0.f};
      if (resid) {
        float4 q0 = ((const float4*)(resid + (size_t)gr * 256))[lane * 2];
        float4 q1 = ((const float4*)(resid + (size_t)gr * 256))[lane * 2 + 1];
        rv[0] = q0.x; rv[1] = q0.y; rv[2] = q0.z; rv[3] = q0.w;
        rv[4] = q1.x; rv[5] = q1.y; rv[6] = q1.z; rv[7] = q1.w;
      }
      float o[8];
      uint4 oh; unsigned short* ou = (unsigned short*)&oh;
#pragma unroll
      for (int i = 0; i < 8; ++i) { o[i] = v[i] + rv[i]; ou[i] = f2bf(o[i]); }
      float4 f0 = {o[0], o[1], o[2], o[3]};
      float4 f1 = {o[4], o[5], o[6], o[7]};
      ((float4*)(outF + (size_t)gr * 256))[lane * 2]     = f0;
      ((float4*)(outF + (size_t)gr * 256))[lane * 2 + 1] = f1;
      ((uint4*)(outH + (size_t)gr * 256))[lane] = oh;
    } else {  // EPI_BF16
      uint4 o; unsigned short* ou = (unsigned short*)&o;
#pragma unroll
      for (int i = 0; i < 8; ++i) ou[i] = f2bf(v[i]);
      ((uint4*)(outH + (size_t)gr * 256))[lane] = o;
    }
  }
}

// ---------------------------------------------------------------------------
// Fused edge MLP: rows = B*E, A = [x[dst] || x[src] || edge_attr] gathered
// straight from global per lane; same 2x2-tile wave layout as the generic
// GEMM. Barrier-free main loop; epilogue: bias -> LN -> SiLU -> atomic add.
// ---------------------------------------------------------------------------
__global__ __launch_bounds__(256) void edge_gemm_kernel(
    const unsigned short* __restrict__ xh, const float* __restrict__ eattr,
    const int* __restrict__ eidx, const unsigned short* __restrict__ Wt, // [256x544]
    const float* __restrict__ bias, const float* __restrict__ gamma,
    const float* __restrict__ beta, float* __restrict__ agg)
{
  __shared__ float sOut[32 * 256];
  const int tid  = threadIdx.x;
  const int wave = tid >> 5, lane = tid & 31;
  const int half = lane >> 4, l16 = lane & 15;
  const int n0 = wave * 16 + l16;
  const int n1 = n0 + 128;
  const long row0 = (long)blockIdx.x * 32;

  // per-lane edge rows for the two 16-row tiles (E = 65536)
  const long g0 = row0 + l16, g1 = row0 + 16 + l16;
  const int b0 = (int)(g0 >> 16), e0 = (int)(g0 & 65535);
  const int b1 = (int)(g1 >> 16), e1 = (int)(g1 & 65535);
  const unsigned short* d0 = xh + ((size_t)b0 * NM_ + eidx[E_ + e0]) * 256 + half * 8;
  const unsigned short* s0 = xh + ((size_t)b0 * NM_ + eidx[e0])      * 256 + half * 8;
  const unsigned short* d1 = xh + ((size_t)b1 * NM_ + eidx[E_ + e1]) * 256 + half * 8;
  const unsigned short* s1 = xh + ((size_t)b1 * NM_ + eidx[e1])      * 256 + half * 8;
  const unsigned short* wRow0 = Wt + (size_t)n0 * 544 + half * 8;
  const unsigned short* wRow1 = Wt + (size_t)n1 * 544 + half * 8;

  v8f a00 = {0.f,0.f,0.f,0.f,0.f,0.f,0.f,0.f};
  v8f a01 = a00, a10 = a00, a11 = a00;
  // region 1: k in [0,256) -> x[dst]
#pragma unroll
  for (int kc = 0; kc < 256; kc += 32) {
    v16bf av0, av1, bv0, bv1;
    ld_frag_a(&av0, d0 + kc);
    ld_frag_a(&av1, d1 + kc);
    ld_frag_b(&bv0, wRow0 + kc);
    ld_frag_b(&bv1, wRow1 + kc);
    a00 = wmma_bf16(av0, bv0, a00);
    a01 = wmma_bf16(av0, bv1, a01);
    a10 = wmma_bf16(av1, bv0, a10);
    a11 = wmma_bf16(av1, bv1, a11);
  }
  // region 2: k in [256,512) -> x[src]
#pragma unroll
  for (int kc = 0; kc < 256; kc += 32) {
    v16bf av0, av1, bv0, bv1;
    ld_frag_a(&av0, s0 + kc);
    ld_frag_a(&av1, s1 + kc);
    ld_frag_b(&bv0, wRow0 + 256 + kc);
    ld_frag_b(&bv1, wRow1 + 256 + kc);
    a00 = wmma_bf16(av0, bv0, a00);
    a01 = wmma_bf16(av0, bv1, a01);
    a10 = wmma_bf16(av1, bv0, a10);
    a11 = wmma_bf16(av1, bv1, a11);
  }
  // region 3: k in [512,544) -> edge_attr (k=512..515) then zeros
  {
    v16bf av0, av1, bv0, bv1;
    uint4 z = {0u, 0u, 0u, 0u};
    ((uint4*)&av0)[0] = z; ((uint4*)&av0)[1] = z;
    ((uint4*)&av1)[0] = z; ((uint4*)&av1)[1] = z;
    if (half == 0) {
      float4 q0 = ((const float4*)eattr)[e0];
      float4 q1 = ((const float4*)eattr)[e1];
      ((unsigned int*)&av0)[0] = f2bf2(q0.x, q0.y);
      ((unsigned int*)&av0)[1] = f2bf2(q0.z, q0.w);
      ((unsigned int*)&av1)[0] = f2bf2(q1.x, q1.y);
      ((unsigned int*)&av1)[1] = f2bf2(q1.z, q1.w);
    }
    ld_frag_b(&bv0, wRow0 + 512);
    ld_frag_b(&bv1, wRow1 + 512);
    a00 = wmma_bf16(av0, bv0, a00);
    a01 = wmma_bf16(av0, bv1, a01);
    a10 = wmma_bf16(av1, bv0, a10);
    a11 = wmma_bf16(av1, bv1, a11);
  }
  const float bb0 = bias[n0], bb1 = bias[n1];
#pragma unroll
  for (int r = 0; r < 8; ++r) {
    int rr0 = (r + half * 8) * 256;
    int rr1 = (16 + r + half * 8) * 256;
    sOut[rr0 + n0] = a00[r] + bb0;
    sOut[rr0 + n1] = a01[r] + bb1;
    sOut[rr1 + n0] = a10[r] + bb0;
    sOut[rr1 + n1] = a11[r] + bb1;
  }
  __syncthreads();

#pragma unroll
  for (int j = 0; j < 4; ++j) {
    int row = wave + 8 * j;
    long gr = row0 + row;
    int b = (int)(gr >> 16);
    int e = (int)(gr & 65535);
    int de = eidx[E_ + e];
    float v[8], s = 0.f, s2 = 0.f;
#pragma unroll
    for (int i = 0; i < 8; ++i) {
      v[i] = sOut[row * 256 + lane * 8 + i];
      s += v[i]; s2 += v[i] * v[i];
    }
    s = waveSum(s); s2 = waveSum(s2);
    float mu = s * (1.f / 256.f);
    float var = s2 * (1.f / 256.f) - mu * mu;
    float rs = rsqrtf(var + 1e-5f);
    float* dstp = agg + ((size_t)b * NM_ + de) * 256 + lane * 8;
#pragma unroll
    for (int i = 0; i < 8; ++i) {
      int c = lane * 8 + i;
      float y = (v[i] - mu) * rs * gamma[c] + beta[c];
      atomicAdd(&dstp[i], siluf(y));
    }
  }
}

// ---------------------------------------------------------------------------
// Weighted index pooling (g2m / m2g); lane handles 8 contiguous cols (uint4).
// ---------------------------------------------------------------------------
__global__ __launch_bounds__(256) void gather_pool_kernel(
    const unsigned short* __restrict__ src, const float* __restrict__ wts,
    const int* __restrict__ idx, const float* __restrict__ extra,
    unsigned short* __restrict__ out, int R, int NIdx, int KI, int Kp, int XD,
    int srcRowsPerB)
{
  const int wave = threadIdx.x >> 5, lane = threadIdx.x & 31;
  long r = (long)blockIdx.x * 8 + wave;
  if (r >= R) return;
  int b = (int)(r / NIdx), m = (int)(r % NIdx);
  float acc[8] = {0.f,0.f,0.f,0.f,0.f,0.f,0.f,0.f};
  for (int k = 0; k < KI; ++k) {
    int j = idx[(size_t)m * KI + k];
    float wv = wts[(size_t)m * KI + k];
    const uint4* sp = (const uint4*)(src + ((size_t)b * srcRowsPerB + j) * 256);
    uint4 q = sp[lane];
    const unsigned short* u = (const unsigned short*)&q;
#pragma unroll
    for (int i = 0; i < 8; ++i) acc[i] += wv * bf2f(u[i]);
  }
  uint4 o;
  unsigned short* ou = (unsigned short*)&o;
#pragma unroll
  for (int i = 0; i < 8; ++i) ou[i] = f2bf(acc[i]);
  ((uint4*)(out + (size_t)r * Kp))[lane] = o;
  for (int c = 256 + lane; c < Kp; c += 32) {
    float val = ((c - 256) < XD) ? extra[(size_t)m * XD + (c - 256)] : 0.0f;
    out[(size_t)r * Kp + c] = f2bf(val);
  }
}

// Build node MLP input: [x_bf16 || bf16(agg)] -> [R x 512], 8 cols per thread
__global__ __launch_bounds__(256) void node_concat_kernel(
    const unsigned short* __restrict__ xh, const float* __restrict__ agg,
    unsigned short* __restrict__ out, long nOct /* = B*NM*256/8 */)
{
  long i = (long)blockIdx.x * 256 + threadIdx.x;
  if (i >= nOct) return;
  long r = i >> 5; int c8 = (int)(i & 31);
  const uint4* xp = (const uint4*)(xh + r * 256);
  ((uint4*)(out + r * 512))[c8] = xp[c8];
  const float* ap = agg + r * 256 + c8 * 8;
  uint4 o; unsigned short* ou = (unsigned short*)&o;
#pragma unroll
  for (int t = 0; t < 8; ++t) ou[t] = f2bf(ap[t]);
  ((uint4*)(out + r * 512 + 256))[c8] = o;
}

// Final decoder: y[r,0:2] = h[r,:] @ dec_w2(256x2) + b2
__global__ __launch_bounds__(256) void dec_out_kernel(
    const unsigned short* __restrict__ h, const float* __restrict__ w2,
    const float* __restrict__ b2, float* __restrict__ y, int R)
{
  const int wave = threadIdx.x >> 5, lane = threadIdx.x & 31;
  long r = (long)blockIdx.x * 8 + wave;
  if (r >= R) return;
  uint4 q = ((const uint4*)(h + (size_t)r * 256))[lane];
  const unsigned short* u = (const unsigned short*)&q;
  float a0 = 0.f, a1 = 0.f;
#pragma unroll
  for (int i = 0; i < 8; ++i) {
    int k = lane * 8 + i;
    float hv = bf2f(u[i]);
    a0 += hv * w2[k * 2 + 0];
    a1 += hv * w2[k * 2 + 1];
  }
  a0 = waveSum(a0); a1 = waveSum(a1);
  if (lane == 0) {
    y[r * 2 + 0] = a0 + b2[0];
    y[r * 2 + 1] = a1 + b2[1];
  }
}

// ---------------------------------------------------------------------------
extern "C" void kernel_launch(void* const* d_in, const int* in_sizes, int n_in,
                              void* d_out, int out_size, void* d_ws, size_t ws_size,
                              hipStream_t stream)
{
  (void)in_sizes; (void)n_in; (void)out_size; (void)ws_size;
  const float* grid_input    = (const float*)d_in[0];
  const float* mesh_features = (const float*)d_in[1];
  const float* edge_attr     = (const float*)d_in[2];
  const float* g2m_weights   = (const float*)d_in[3];
  const float* m2g_weights   = (const float*)d_in[4];
  const int*   edge_index    = (const int*)d_in[5];
  const int*   g2m_indices   = (const int*)d_in[6];
  const int*   m2g_indices   = (const int*)d_in[7];
  const float* grid_w1   = (const float*)d_in[8];
  const float* grid_b1   = (const float*)d_in[9];
  const float* grid_g    = (const float*)d_in[10];
  const float* grid_beta = (const float*)d_in[11];
  const float* grid_w2   = (const float*)d_in[12];
  const float* grid_b2   = (const float*)d_in[13];
  const float* comb_w1   = (const float*)d_in[14];
  const float* comb_b1   = (const float*)d_in[15];
  const float* comb_g    = (const float*)d_in[16];
  const float* comb_beta = (const float*)d_in[17];
  const float* comb_w2   = (const float*)d_in[18];
  const float* comb_b2   = (const float*)d_in[19];
  const float* edge_w    = (const float*)d_in[20];
  const float* edge_b    = (const float*)d_in[21];
  const float* edge_g    = (const float*)d_in[22];
  const float* edge_beta = (const float*)d_in[23];
  const float* node_w1   = (const float*)d_in[24];
  const float* node_b1   = (const float*)d_in[25];
  const float* node_g    = (const float*)d_in[26];
  const float* node_beta = (const float*)d_in[27];
  const float* node_w2   = (const float*)d_in[28];
  const float* node_b2   = (const float*)d_in[29];
  const float* dec_w1    = (const float*)d_in[30];
  const float* dec_b1    = (const float*)d_in[31];
  const float* dec_g     = (const float*)d_in[32];
  const float* dec_beta  = (const float*)d_in[33];
  const float* dec_w2    = (const float*)d_in[34];
  const float* dec_b2    = (const float*)d_in[35];
  float* y_out = (float*)d_out;

  const long RG = (long)B_ * NG_;   // 163848
  const long RM = (long)B_ * NM_;   //  40968
  const long RE = (long)B_ * E_;    // 262144

  char* ws = (char*)d_ws;
  size_t off = 0;
  auto alloc = [&](size_t bytes) -> void* {
    void* p = ws + off;
    off = (off + bytes + 255) & ~(size_t)255;
    return p;
  };
  unsigned short* bufA_ng = (unsigned short*)alloc((size_t)RG * 256 * 2);
  unsigned short* bufB_ng = (unsigned short*)alloc((size_t)RG * 256 * 2);
  unsigned short* nodeA   = (unsigned short*)alloc((size_t)RM * 512 * 2);
  unsigned short* hid_nm  = (unsigned short*)alloc((size_t)RM * 256 * 2);
  float*          x_f32   = (float*)alloc((size_t)RM * 256 * 4);
  unsigned short* x_bf16  = (unsigned short*)alloc((size_t)RM * 256 * 2);
  float*          agg     = (float*)alloc((size_t)RM * 256 * 4);
  // pre-converted bf16 transposed weights
  unsigned short* wt_grid2 = (unsigned short*)alloc(256 * 256 * 2);
  unsigned short* wt_comb1 = (unsigned short*)alloc(256 * 288 * 2);
  unsigned short* wt_comb2 = (unsigned short*)alloc(256 * 256 * 2);
  unsigned short* wt_edge  = (unsigned short*)alloc((size_t)NL_ * 256 * 544 * 2);
  unsigned short* wt_node1 = (unsigned short*)alloc((size_t)NL_ * 256 * 512 * 2);
  unsigned short* wt_node2 = (unsigned short*)alloc((size_t)NL_ * 256 * 256 * 2);
  unsigned short* wt_dec1  = (unsigned short*)alloc(256 * 256 * 2);

  auto convW = [&](const float* W, unsigned short* Wt, int K, int KP) {
    long total = 128L * KP;  // 256*KP/2
    wt_convert_kernel<<<(int)((total + 255) / 256), 256, 0, stream>>>(
        W, Wt, K, KP, total);
  };
  convW(grid_w2, wt_grid2, 256, 256);
  convW(comb_w1, wt_comb1, 259, 288);
  convW(comb_w2, wt_comb2, 256, 256);
  for (int l = 0; l < NL_; ++l) {
    convW(edge_w  + (size_t)l * 516 * 256, wt_edge  + (size_t)l * 256 * 544, 516, 544);
    convW(node_w1 + (size_t)l * 512 * 256, wt_node1 + (size_t)l * 256 * 512, 512, 512);
    convW(node_w2 + (size_t)l * 256 * 256, wt_node2 + (size_t)l * 256 * 256, 256, 256);
  }
  convW(dec_w1, wt_dec1, 256, 256);

  const int gemmBlkG = (int)((RG + 31) / 32);
  const int gemmBlkM = (int)((RM + 31) / 32);
  const int rowBlkG  = (int)((RG + 7) / 8);
  const int rowBlkM  = (int)((RM + 7) / 8);

  // --- Grid encoder ---
  encode_kernel<<<rowBlkG, 256, 0, stream>>>(grid_input, grid_w1, grid_b1,
                                             grid_g, grid_beta, bufA_ng, (int)RG);
  gemm256_kernel<256, EPI_BF16><<<gemmBlkG, 256, 0, stream>>>(
      bufA_ng, wt_grid2, grid_b2, nullptr, nullptr, nullptr, nullptr, bufB_ng,
      (int)RG);

  // --- g2m pooling + mesh_features concat (KP=288) ---
  gather_pool_kernel<<<rowBlkM, 256, 0, stream>>>(bufB_ng, g2m_weights,
      g2m_indices, mesh_features, nodeA, (int)RM, NM_, 16, 288, 3, NG_);
  gemm256_kernel<288, EPI_LNSILU><<<gemmBlkM, 256, 0, stream>>>(
      nodeA, wt_comb1, comb_b1, comb_g, comb_beta, nullptr, nullptr, hid_nm,
      (int)RM);
  gemm256_kernel<256, EPI_X><<<gemmBlkM, 256, 0, stream>>>(
      hid_nm, wt_comb2, comb_b2, nullptr, nullptr, nullptr, x_f32, x_bf16,
      (int)RM);

  // --- Message-passing layers ---
  for (int l = 0; l < NL_; ++l) {
    hipMemsetAsync(agg, 0, (size_t)RM * 256 * 4, stream);
    edge_gemm_kernel<<<(int)(RE / 32), 256, 0, stream>>>(x_bf16, edge_attr,
        edge_index, wt_edge + (size_t)l * 256 * 544, edge_b + (size_t)l * 256,
        edge_g + (size_t)l * 256, edge_beta + (size_t)l * 256, agg);
    long nOct = RM * 32;    // R * 256 / 8
    node_concat_kernel<<<(int)((nOct + 255) / 256), 256, 0, stream>>>(
        x_bf16, agg, nodeA, nOct);
    gemm256_kernel<512, EPI_LNSILU><<<gemmBlkM, 256, 0, stream>>>(
        nodeA, wt_node1 + (size_t)l * 256 * 512, node_b1 + (size_t)l * 256,
        node_g + (size_t)l * 256, node_beta + (size_t)l * 256,
        nullptr, nullptr, hid_nm, (int)RM);
    gemm256_kernel<256, EPI_X><<<gemmBlkM, 256, 0, stream>>>(
        hid_nm, wt_node2 + (size_t)l * 256 * 256, node_b2 + (size_t)l * 256,
        nullptr, nullptr, x_f32, x_f32, x_bf16, (int)RM);
  }

  // --- m2g pooling + decoder ---
  gather_pool_kernel<<<rowBlkG, 256, 0, stream>>>(x_bf16, m2g_weights,
      m2g_indices, nullptr, bufA_ng, (int)RG, NG_, 4, 256, 0, NM_);
  gemm256_kernel<256, EPI_LNSILU><<<gemmBlkG, 256, 0, stream>>>(
      bufA_ng, wt_dec1, dec_b1, dec_g, dec_beta, nullptr, nullptr, bufB_ng,
      (int)RG);
  dec_out_kernel<<<rowBlkG, 256, 0, stream>>>(bufB_ng, dec_w2, dec_b2,
                                              y_out, (int)RG);
}
